// GraphUpdater_90692529422343
// MI455X (gfx1250) — matrix operand order repaired and checked by reading, not compiled, not run on hardware
//
#include <hip/hip_runtime.h>
#include <hip/hip_bf16.h>

typedef __attribute__((ext_vector_type(16))) _Float16 v16h;
typedef __attribute__((ext_vector_type(8)))  float    v8f;
typedef unsigned int u32x4 __attribute__((ext_vector_type(4)));
typedef int          i32x8 __attribute__((ext_vector_type(8)));
typedef int          i32x4 __attribute__((ext_vector_type(4)));

#define B_  16
#define N_  1024
#define R_  10
#define DE_ 64
#define DR_ 32
#define H_  64
#define CAT_ 960   // R_ * 96

// ---------------------------------------------------------------------------
// Prep: zT[b][r][h][n] = sum_d X[b][n][d] * W[h][r*96+d] + c_{b,r}[h]
//       c_{b,r}[h]     = sum_d rel[b][r][d] * W[h][r*96+64+d]
// Stored transposed (h-major, n contiguous) as f16 for the WMMA B operand.
// ---------------------------------------------------------------------------
__global__ __launch_bounds__(256) void rgcn_prep(
    const float* __restrict__ X,      // [B,N,64] f32
    const float* __restrict__ rel,    // [B,R,32] f32
    const float* __restrict__ W,      // [64,960] f32
    _Float16* __restrict__ zT)        // [B,R,64,N] f16
{
    __shared__ float Wl[64 * 96];
    __shared__ float cL[64];

    const int t      = threadIdx.x;
    const int blk    = blockIdx.x;          // B*R*8 blocks
    const int nchunk = blk & 7;
    const int r      = (blk >> 3) % R_;
    const int b      = blk / (8 * R_);

    // stage W[:, r*96 : (r+1)*96] into LDS
    for (int i = t; i < 64 * 96; i += 256) {
        int h = i / 96, d = i - h * 96;
        Wl[i] = W[h * CAT_ + r * 96 + d];
    }
    __syncthreads();

    if (t < 64) {
        const float* rp = rel + ((size_t)b * R_ + r) * DR_;
        float acc = 0.f;
        for (int d = 0; d < DR_; ++d) acc += rp[d] * Wl[t * 96 + 64 + d];
        cL[t] = acc;
    }
    __syncthreads();

    const int n     = nchunk * 128 + (t & 127);
    const int hbase = t >> 7;               // 0 or 1

    const float* xr = X + ((size_t)b * N_ + n) * DE_;
    float4 xv[16];
#pragma unroll
    for (int i = 0; i < 16; ++i) xv[i] = ((const float4*)xr)[i];

    _Float16* zbase = zT + (((size_t)b * R_ + r) * H_) * N_ + n;
#pragma unroll 2
    for (int h = hbase; h < 64; h += 2) {
        float acc = cL[h];
        const float4* wrow = (const float4*)&Wl[h * 96];
#pragma unroll
        for (int i = 0; i < 16; ++i) {
            float4 w = wrow[i];
            acc += xv[i].x * w.x + xv[i].y * w.y + xv[i].z * w.z + xv[i].w * w.w;
        }
        zbase[(size_t)h * N_] = (_Float16)acc;
    }
}

// ---------------------------------------------------------------------------
// Main: out[b][m][h] = relu( sum_r sum_n adj[b,r,m,n] * z[b,r,n,h] + bias[h] )
// 4 waves / 128 threads per WG; wave owns a 16-row m-strip x 64 h-cols.
// z staged via Tensor Data Mover (tensor_load_to_lds, TENSORcnt), one 4KB
// DMA per wave per 128-wide K block; A = adj f32 -> f16 converted in-register.
// ---------------------------------------------------------------------------
__global__ __launch_bounds__(128) void rgcn_main(
    const float* __restrict__ adj,       // [B,R,N,N] f32
    const _Float16* __restrict__ zT,     // [B,R,64,N] f16
    const float* __restrict__ bias,      // [64] f32
    float* __restrict__ out)             // [B,N,64] f32
{
    constexpr int LSTR = 132;            // halves per LDS row (128 + 8B TDM pad)
    __shared__ _Float16 ldsZ[64 * LSTR]; // 16896 B

    const int t    = threadIdx.x;
    const int lane = t & 31;
    const int c    = lane & 15;          // column / M-row index within frags
    const int g    = lane >> 4;          // lane group (0/1)

    // wave id, made explicitly scalar for the TDM descriptor
    const int wv = __builtin_amdgcn_readfirstlane(t >> 5);

    const int b  = blockIdx.x >> 4;      // 16 m-blocks of 64 rows per batch
    const int m0 = ((blockIdx.x & 15) << 6) + (wv << 4);
    const int row = m0 + c;              // A-matrix row this lane loads

    v8f acc[4];
#pragma unroll
    for (int i = 0; i < 4; ++i) acc[i] = (v8f){0.f,0.f,0.f,0.f,0.f,0.f,0.f,0.f};

    // ---- TDM descriptor (D#): 2-D tile [16 h-rows x 128 n-cols], f16,
    //      pad 2 DWORDs after every 64 DWORDs -> LDS row stride 264 B ----
    const unsigned ldsBase =
        (unsigned)(uintptr_t)&ldsZ[0] + (unsigned)(16 * wv) * (LSTR * 2);

    i32x8 g1;
    g1[0] = (1 << 16)        // data_size = 2 bytes
          | (1 << 20)        // pad_enable
          | (5 << 22)        // pad_interval: 64 DWORDs
          | (1 << 25);       // pad_amount: 2 DWORDs
    g1[1] = (int)(1024u << 16);   // tensor_dim0 = 1024 (bits 79:48, low half)
    g1[2] = (int)(64u   << 16);   // tensor_dim1 = 64   (bits 111:80, low half)
    g1[3] = (int)(128u  << 16);   // tile_dim0  = 128   (bits 127:112)
    g1[4] = 16;                   // tile_dim1  = 16
    g1[5] = 1024;                 // tensor_dim0_stride = 1024 elements
    g1[6] = 0;
    g1[7] = 0;
    const i32x4 gz4 = {0, 0, 0, 0};            // groups 2/3 unused (2-D tensor)
    const i32x8 gz8 = {0, 0, 0, 0, 0, 0, 0, 0};

    const unsigned long long zbytes = (unsigned long long)(uintptr_t)zT;

    for (int r = 0; r < R_; ++r) {
        const float* adjRow =
            adj + ((((size_t)b * R_ + r) * N_) + row) * N_;
        const unsigned long long gwave =
            zbytes + 2ull * ((((unsigned long long)b * R_ + r) * H_ + 16 * wv) * N_);

        for (int nb = 0; nb < N_; nb += 128) {
            // ---- issue TDM: zT[b][r][16wv..16wv+15][nb..nb+127] -> LDS ----
            const unsigned long long ga = gwave + 2ull * nb;
            u32x4 g0;
            g0[0] = 1u;                                   // count = 1
            g0[1] = ldsBase;                              // lds_addr
            g0[2] = (unsigned)(ga & 0xFFFFFFFFu);         // global_addr lo
            g0[3] = (unsigned)((ga >> 32) & 0x01FFFFFFu)  // global_addr hi
                  | (2u << 30);                           // type = 2 (image)
            __syncthreads();   // prior rounds' LDS reads done
            __builtin_amdgcn_tensor_load_to_lds(g0, g1, gz4, gz4, gz8, 0);
            __builtin_amdgcn_s_wait_tensorcnt(0);
            __syncthreads();   // all 4 wave-quarters visible

            // ---- 4 K-steps of 32 over this 128-wide block ----
#pragma unroll
            for (int k = 0; k < 4; ++k) {
                const int n0 = nb + 32 * k;

                // A fragment: adj[row, n0..n0+31] f32 -> f16, ISA layout:
                // lane group g: VGPR0-3 = K 8g..8g+7, VGPR4-7 = K 16+8g..+7
                union { v16h v; _Float16 e[16]; } A;
                {
                    const float* p = adjRow + n0 + 8 * g;
                    float4 a0 = ((const float4*)p)[0];
                    float4 a1 = ((const float4*)p)[1];
                    const float* q = adjRow + n0 + 16 + 8 * g;
                    float4 a2 = ((const float4*)q)[0];
                    float4 a3 = ((const float4*)q)[1];
                    A.e[0]  = (_Float16)a0.x; A.e[1]  = (_Float16)a0.y;
                    A.e[2]  = (_Float16)a0.z; A.e[3]  = (_Float16)a0.w;
                    A.e[4]  = (_Float16)a1.x; A.e[5]  = (_Float16)a1.y;
                    A.e[6]  = (_Float16)a1.z; A.e[7]  = (_Float16)a1.w;
                    A.e[8]  = (_Float16)a2.x; A.e[9]  = (_Float16)a2.y;
                    A.e[10] = (_Float16)a2.z; A.e[11] = (_Float16)a2.w;
                    A.e[12] = (_Float16)a3.x; A.e[13] = (_Float16)a3.y;
                    A.e[14] = (_Float16)a3.z; A.e[15] = (_Float16)a3.w;
                }

                // 4 h-tiles: B frag = K 8g..8g+7, 16+8g..+7 halves from padded LDS
#pragma unroll
                for (int ht = 0; ht < 4; ++ht) {
                    union { v16h v; unsigned long long q[4]; } Bf;
                    const unsigned long long* bl = (const unsigned long long*)
                        &ldsZ[(ht * 16 + c) * LSTR + 32 * k + 8 * g];
                    Bf.q[0] = bl[0]; Bf.q[1] = bl[1];
                    Bf.q[2] = bl[4]; Bf.q[3] = bl[5];
                    acc[ht] = __builtin_amdgcn_wmma_f32_16x16x32_f16(
                        false, A.v, false, Bf.v, (short)0, acc[ht],
                        false, false);
                }
            }
        }
    }

    // ---- epilogue: bias + ReLU; C/D layout: M = v + 8g, N = c ----
#pragma unroll
    for (int ht = 0; ht < 4; ++ht) {
        const int h  = ht * 16 + c;
        const float bv = bias[h];
#pragma unroll
        for (int v = 0; v < 8; ++v) {
            const int m = m0 + 8 * g + v;
            float val = acc[ht][v] + bv;
            out[(((size_t)b) * N_ + m) * H_ + h] = val > 0.f ? val : 0.f;
        }
    }
}

// ---------------------------------------------------------------------------
extern "C" void kernel_launch(void* const* d_in, const int* in_sizes, int n_in,
                              void* d_out, int out_size, void* d_ws, size_t ws_size,
                              hipStream_t stream) {
    (void)in_sizes; (void)n_in; (void)out_size; (void)ws_size;
    const float* X   = (const float*)d_in[0];   // [16,1024,64]
    const float* rel = (const float*)d_in[1];   // [16,10,32]
    const float* adj = (const float*)d_in[2];   // [16,10,1024,1024]
    const float* W0  = (const float*)d_in[3];   // [64,960]
    const float* b0  = (const float*)d_in[4];   // [64]
    const float* W1  = (const float*)d_in[5];   // [64,960]
    const float* b1  = (const float*)d_in[6];   // [64]
    float* out = (float*)d_out;                 // [16,1024,64]

    char* ws = (char*)d_ws;
    _Float16* zT = (_Float16*)ws;                                   // 20 MB
    float*    x1 = (float*)(ws + (size_t)B_ * R_ * H_ * N_ * sizeof(_Float16)); // 4 MB

    const int prepGrid = B_ * R_ * 8;    // 1280
    const int mainGrid = B_ * (N_ / 64); // 256

    // Layer 1
    rgcn_prep<<<prepGrid, 256, 0, stream>>>(X, rel, W0, zT);
    rgcn_main<<<mainGrid, 128, 0, stream>>>(adj, zT, b0, x1);
    // Layer 2
    rgcn_prep<<<prepGrid, 256, 0, stream>>>(x1, rel, W1, zT);
    rgcn_main<<<mainGrid, 128, 0, stream>>>(adj, zT, b1, out);
}
